// HeteroGraphSAGE_5403068858515
// MI455X (gfx1250) — matrix operand order repaired and checked by reading, not compiled, not run on hardware
//
#include <hip/hip_runtime.h>

// ---------------- types ----------------
typedef __attribute__((ext_vector_type(16))) __bf16 v16bf;
typedef __attribute__((ext_vector_type(8)))  __bf16 v8bf;
typedef __attribute__((ext_vector_type(8)))  float  v8f;

#define N_CARD 100000
#define N_TX   1000000
#define N_MER  10000
#define NEDGE  1000000
#define HDIM   128

#if defined(__has_builtin)
#if __has_builtin(__builtin_amdgcn_sched_barrier)
#define SCHED_FENCE() __builtin_amdgcn_sched_barrier(0x7)  // block VMEM+WMMA crossing
#endif
#endif
#ifndef SCHED_FENCE
#define SCHED_FENCE()
#endif

// ---------------- WMMA helpers ----------------
__device__ __forceinline__ v8f wmma_bf16(v16bf a, v16bf b, v8f c) {
  return __builtin_amdgcn_wmma_f32_16x16x32_bf16(false, a, false, b, (short)0, c, false, false);
}

// Build A tile (16x32, MxK) from f32 row-major rows, scaled by s.
// ISA layout: lanes 0-15 (row=lane): elems 0..7 = K 0..7, elems 8..15 = K 16..23
//             lanes 16-31 (row=lane-16): elems 0..7 = K 8..15, 8..15 = K 24..31
__device__ __forceinline__ v16bf a_from_f32(const float* rowptr, int koff, float s) {
  float4 f0 = *(const float4*)(rowptr + koff);
  float4 f1 = *(const float4*)(rowptr + koff + 4);
  float4 g0 = *(const float4*)(rowptr + koff + 16);
  float4 g1 = *(const float4*)(rowptr + koff + 20);
  v16bf a;
  a[0]  = (__bf16)(f0.x * s); a[1]  = (__bf16)(f0.y * s);
  a[2]  = (__bf16)(f0.z * s); a[3]  = (__bf16)(f0.w * s);
  a[4]  = (__bf16)(f1.x * s); a[5]  = (__bf16)(f1.y * s);
  a[6]  = (__bf16)(f1.z * s); a[7]  = (__bf16)(f1.w * s);
  a[8]  = (__bf16)(g0.x * s); a[9]  = (__bf16)(g0.y * s);
  a[10] = (__bf16)(g0.z * s); a[11] = (__bf16)(g0.w * s);
  a[12] = (__bf16)(g1.x * s); a[13] = (__bf16)(g1.y * s);
  a[14] = (__bf16)(g1.z * s); a[15] = (__bf16)(g1.w * s);
  return a;
}

__device__ __forceinline__ v16bf a_from_bf16(const __bf16* rowptr, int koff) {
  v8bf lo = *(const v8bf*)(rowptr + koff);
  v8bf hi = *(const v8bf*)(rowptr + koff + 16);
  v16bf a;
#pragma unroll
  for (int j = 0; j < 8; ++j) { a[j] = lo[j]; a[8 + j] = hi[j]; }
  return a;
}

// B tile (32x16, KxN) from pre-transposed weight Wt stored [N][K] bf16.
// lane n (<16): col N=n, K = kc*32 + 0..7,16..23 ; lane n+16: same col, K = kc*32 + 8..15,24..31
__device__ __forceinline__ v16bf b_tile(const __bf16* Wt, int K, int kc, int nt, int lane) {
  int n = nt * 16 + (lane & 15);
  int kbase = kc * 32 + ((lane >> 4) << 3);
  v8bf lo = *(const v8bf*)(Wt + (long)n * K + kbase);
  v8bf hi = *(const v8bf*)(Wt + (long)n * K + kbase + 16);
  return __builtin_shufflevector(lo, hi, 0, 1, 2, 3, 4, 5, 6, 7, 8, 9, 10, 11, 12, 13, 14, 15);
}

// Preload 8 B tiles (one K-chunk across the full N=128) into distinct registers,
// then fire 8 back-to-back WMMAs. The sched_barrier keeps all 16 b128 loads live
// across the fence so the RA cannot recycle registers -> one clause of loads,
// one s_wait_loadcnt, 8 WMMAs (next chunk's loads overlap the XDL pipe).
__device__ __forceinline__ void kchunk_mma(v8f* acc, v16bf a, const __bf16* Wt, int K,
                                           int kc, int lane) {
  v16bf B[8];
#pragma unroll
  for (int nt = 0; nt < 8; ++nt) B[nt] = b_tile(Wt, K, kc, nt, lane);
  SCHED_FENCE();
#pragma unroll
  for (int nt = 0; nt < 8; ++nt) acc[nt] = wmma_bf16(a, B[nt], acc[nt]);
  SCHED_FENCE();
}

// Epilogue: bias + relu + bf16 store, row-major [node][128].
// C layout: lane = col (lane&15) of N-subtile, vgpr r = row r (+8 for lanes>=16).
__device__ __forceinline__ void store_relu_bf16(const v8f* acc, const float* bias,
                                                __bf16* out, long r0, int lane) {
  int col0 = lane & 15;
  int mh = (lane >= 16) ? 8 : 0;
#pragma unroll
  for (int nt = 0; nt < 8; ++nt) {
    float bv = bias[nt * 16 + col0];
#pragma unroll
    for (int r = 0; r < 8; ++r) {
      float v = fmaxf(acc[nt][r] + bv, 0.0f);
      out[(r0 + mh + r) * HDIM + nt * 16 + col0] = (__bf16)v;
    }
  }
}

// ---------------- kernel 0: weight prep (transpose + bf16 + fuse) ----------------
__global__ void prep_weights(const float* pays_wl, const float* pays_bl, const float* pays_wr,
                             const float* recv_wl, const float* recv_bl, const float* recv_wr,
                             const float* rpays_wl, const float* rpays_wr,
                             const float* rrecv_wl, const float* rrecv_wr,
                             const float* w2l, const float* b2, const float* w2r,
                             __bf16* Wt_pl, __bf16* Wt_rl, __bf16* Wt_txr,
                             __bf16* Wt_rp_l, __bf16* Wt_rp_r,
                             __bf16* Wt_rr_l, __bf16* Wt_rr_r,
                             __bf16* Wt2c, __bf16* Wt2m, __bf16* Wt2t,
                             float* b1tx, float* b2c) {
  int n = blockIdx.x;   // output feature 0..127
  int k = threadIdx.x;  // input feature 0..127
  if (k < 32) {
    Wt_pl[n * 32 + k]   = (__bf16)pays_wl[k * HDIM + n];
    Wt_rl[n * 32 + k]   = (__bf16)recv_wl[k * HDIM + n];
    Wt_rp_r[n * 32 + k] = (__bf16)rpays_wr[k * HDIM + n];
    Wt_rr_r[n * 32 + k] = (__bf16)rrecv_wr[k * HDIM + n];
  }
  if (k < 64) {
    Wt_txr[n * 64 + k]  = (__bf16)(pays_wr[k * HDIM + n] + recv_wr[k * HDIM + n]);
    Wt_rp_l[n * 64 + k] = (__bf16)rpays_wl[k * HDIM + n];
    Wt_rr_l[n * 64 + k] = (__bf16)rrecv_wl[k * HDIM + n];
  }
  Wt2c[n * HDIM + k] = (__bf16)w2l[0 * HDIM * HDIM + k * HDIM + n];
  Wt2m[n * HDIM + k] = (__bf16)w2l[1 * HDIM * HDIM + k * HDIM + n];
  Wt2t[n * HDIM + k] = (__bf16)(w2r[0 * HDIM * HDIM + k * HDIM + n] +
                                w2r[1 * HDIM * HDIM + k * HDIM + n]);
  if (k == 0) {
    b1tx[n] = pays_bl[n] + recv_bl[n];
    b2c[n]  = b2[n] + b2[HDIM + n];
  }
}

// ---------------- kernel 1: segment-sum scatter (reverse relations) ----------------
__global__ void scatter_tx(const float* x_tx, const int* ps, const int* pd,
                           const int* rs, const int* rd,
                           float* card_sum, float* cnt_card,
                           float* mer_sum, float* cnt_mer) {
  long gid = (long)blockIdx.x * blockDim.x + threadIdx.x;
  long e = gid >> 4;          // edge
  int ch = (int)(gid & 15);   // 4-float chunk of the 64-dim row
  if (e >= NEDGE) return;
  int cdst = ps[e];
  int mdst = rs[e];
  int trc = pd[e];
  int trm = rd[e];
  float4 vc = *(const float4*)(x_tx + (long)trc * 64 + ch * 4);
  float4 vm = (trm == trc) ? vc : *(const float4*)(x_tx + (long)trm * 64 + ch * 4);
  float* cp = card_sum + (long)cdst * 64 + ch * 4;
  atomicAdd(cp + 0, vc.x); atomicAdd(cp + 1, vc.y);
  atomicAdd(cp + 2, vc.z); atomicAdd(cp + 3, vc.w);
  float* mp = mer_sum + (long)mdst * 64 + ch * 4;
  atomicAdd(mp + 0, vm.x); atomicAdd(mp + 1, vm.y);
  atomicAdd(mp + 2, vm.z); atomicAdd(mp + 3, vm.w);
  if (ch == 0) {
    atomicAdd(cnt_card + cdst, 1.0f);
    atomicAdd(cnt_mer + mdst, 1.0f);
  }
}

// ---------------- kernel 2: layer-1 node update (card / mer) ----------------
// out[c] = relu( (sum[c]/max(cnt,1)) @ Wl + x[c] @ Wr + b ),  sum K=64, x K=32
__global__ void node_l1(const float* sum, const float* cnt, const float* x,
                        const __bf16* Wt_l, const __bf16* Wt_r, const float* bias,
                        __bf16* out, int nNodes) {
  int lane = threadIdx.x & 31;
  int tile = blockIdx.x * (blockDim.x >> 5) + (threadIdx.x >> 5);
  int ntiles = nNodes >> 4;
  if (tile >= ntiles) return;
  long r0 = (long)tile * 16;
  int row = lane & 15;
  int koff = (lane >> 4) * 8;
  float c = cnt[r0 + row];
  float inv = 1.0f / fmaxf(c, 1.0f);
  const float* srow = sum + (r0 + row) * 64;
  const float* xrow = x + (r0 + row) * 32;
  v8f acc[8];
#pragma unroll
  for (int nt = 0; nt < 8; ++nt) acc[nt] = (v8f){0, 0, 0, 0, 0, 0, 0, 0};
#pragma unroll
  for (int kc = 0; kc < 2; ++kc) {
    v16bf a = a_from_f32(srow + kc * 32, koff, inv);
    kchunk_mma(acc, a, Wt_l, 64, kc, lane);
  }
  {
    v16bf a = a_from_f32(xrow, koff, 1.0f);
    kchunk_mma(acc, a, Wt_r, 32, 0, lane);
  }
  store_relu_bf16(acc, bias, out, r0, lane);
}

// ---------------- kernel 3: layer-1 transaction update ----------------
// tx1[i] = relu( x_card[ps[i]]@Wpl + x_mer[rs[i]]@Wrl + x_tx[i]@Wtxr + b )
__global__ void tx_l1(const float* x_card, const float* x_mer, const float* x_tx,
                      const int* ps, const int* rs,
                      const __bf16* Wt_pl, const __bf16* Wt_rl, const __bf16* Wt_txr,
                      const float* bias, __bf16* out) {
  int lane = threadIdx.x & 31;
  int tile = blockIdx.x * (blockDim.x >> 5) + (threadIdx.x >> 5);
  long r0 = (long)tile * 16;
  int row = lane & 15;
  int koff = (lane >> 4) * 8;
  int ci = ps[r0 + row];
  int mi = rs[r0 + row];
  const float* crow = x_card + (long)ci * 32;
  const float* mrow = x_mer + (long)mi * 32;
  const float* trow = x_tx + (r0 + row) * 64;
  v8f acc[8];
#pragma unroll
  for (int nt = 0; nt < 8; ++nt) acc[nt] = (v8f){0, 0, 0, 0, 0, 0, 0, 0};
  {
    v16bf a = a_from_f32(crow, koff, 1.0f);
    kchunk_mma(acc, a, Wt_pl, 32, 0, lane);
  }
  {
    v16bf a = a_from_f32(mrow, koff, 1.0f);
    kchunk_mma(acc, a, Wt_rl, 32, 0, lane);
  }
#pragma unroll
  for (int kc = 0; kc < 2; ++kc) {
    v16bf a = a_from_f32(trow + kc * 32, koff, 1.0f);
    kchunk_mma(acc, a, Wt_txr, 64, kc, lane);
  }
  store_relu_bf16(acc, bias, out, r0, lane);
}

// ---------------- kernel 4: layer-2 transaction + fused head ----------------
// out[i] = relu( card1[ps[i]]@W2c + mer1[rs[i]]@W2m + tx1[i]@W2t + b2c ) . wf + bf
__global__ void tx_l2_head(const __bf16* card1, const __bf16* mer1, const __bf16* tx1,
                           const int* ps, const int* rs,
                           const __bf16* Wt2c, const __bf16* Wt2m, const __bf16* Wt2t,
                           const float* b2c, const float* wf, const float* bfp,
                           float* out) {
  int lane = threadIdx.x & 31;
  int tile = blockIdx.x * (blockDim.x >> 5) + (threadIdx.x >> 5);
  long r0 = (long)tile * 16;
  int row = lane & 15;
  int koff = (lane >> 4) * 8;
  int ci = ps[r0 + row];
  int mi = rs[r0 + row];
  const __bf16* crow = card1 + (long)ci * HDIM;
  const __bf16* mrow = mer1 + (long)mi * HDIM;
  const __bf16* trow = tx1 + (r0 + row) * HDIM;
  v8f acc[8];
#pragma unroll
  for (int nt = 0; nt < 8; ++nt) acc[nt] = (v8f){0, 0, 0, 0, 0, 0, 0, 0};
#pragma unroll
  for (int kc = 0; kc < 4; ++kc) {
    v16bf a = a_from_bf16(crow + kc * 32, koff);
    kchunk_mma(acc, a, Wt2c, HDIM, kc, lane);
  }
#pragma unroll
  for (int kc = 0; kc < 4; ++kc) {
    v16bf a = a_from_bf16(mrow + kc * 32, koff);
    kchunk_mma(acc, a, Wt2m, HDIM, kc, lane);
  }
#pragma unroll
  for (int kc = 0; kc < 4; ++kc) {
    v16bf a = a_from_bf16(trow + kc * 32, koff);
    kchunk_mma(acc, a, Wt2t, HDIM, kc, lane);
  }
  // fused head: bias + relu + dot with wf, cross-lane reduce over the 16-lane half
  float p[8];
#pragma unroll
  for (int r = 0; r < 8; ++r) p[r] = 0.0f;
  int col0 = lane & 15;
#pragma unroll
  for (int nt = 0; nt < 8; ++nt) {
    float bv = b2c[nt * 16 + col0];
    float wv = wf[nt * 16 + col0];
#pragma unroll
    for (int r = 0; r < 8; ++r)
      p[r] += fmaxf(acc[nt][r] + bv, 0.0f) * wv;
  }
#pragma unroll
  for (int off = 8; off >= 1; off >>= 1) {
#pragma unroll
    for (int r = 0; r < 8; ++r)
      p[r] += __shfl_xor(p[r], off, 32);
  }
  float bfv = bfp[0];
  int mh = (lane >= 16) ? 8 : 0;
#pragma unroll
  for (int r = 0; r < 8; ++r)
    if (col0 == r) out[r0 + mh + r] = p[r] + bfv;
}

// ---------------- host launcher ----------------
extern "C" void kernel_launch(void* const* d_in, const int* in_sizes, int n_in,
                              void* d_out, int out_size, void* d_ws, size_t ws_size,
                              hipStream_t stream) {
  (void)in_sizes; (void)n_in; (void)out_size; (void)ws_size;
  const float* x_card      = (const float*)d_in[0];
  const float* x_tx        = (const float*)d_in[1];
  const float* x_mer       = (const float*)d_in[2];
  const float* l1_pays_wl  = (const float*)d_in[3];
  const float* l1_pays_bl  = (const float*)d_in[4];
  const float* l1_pays_wr  = (const float*)d_in[5];
  const float* l1_recv_wl  = (const float*)d_in[6];
  const float* l1_recv_bl  = (const float*)d_in[7];
  const float* l1_recv_wr  = (const float*)d_in[8];
  const float* l1_rpays_wl = (const float*)d_in[9];
  const float* l1_rpays_bl = (const float*)d_in[10];
  const float* l1_rpays_wr = (const float*)d_in[11];
  const float* l1_rrecv_wl = (const float*)d_in[12];
  const float* l1_rrecv_bl = (const float*)d_in[13];
  const float* l1_rrecv_wr = (const float*)d_in[14];
  const float* w2l = (const float*)d_in[15];
  const float* b2  = (const float*)d_in[16];
  const float* w2r = (const float*)d_in[17];
  const float* wf  = (const float*)d_in[18];
  const float* bf  = (const float*)d_in[19];
  const int* ps = (const int*)d_in[20];
  const int* pd = (const int*)d_in[21];
  const int* rs = (const int*)d_in[22];
  const int* rd = (const int*)d_in[23];

  // workspace carving (256B aligned)
  char* w = (char*)d_ws;
  size_t off = 0;
  auto carve = [&](size_t bytes) -> char* {
    char* p = w + off;
    off = (off + bytes + 255) & ~(size_t)255;
    return p;
  };
  float* card_sum = (float*)carve((size_t)N_CARD * 64 * 4);
  float* mer_sum  = (float*)carve((size_t)N_MER * 64 * 4);
  float* cnt_card = (float*)carve((size_t)N_CARD * 4);
  float* cnt_mer  = (float*)carve((size_t)N_MER * 4);
  size_t zero_bytes = off;  // zero everything up to here
  __bf16* tx1   = (__bf16*)carve((size_t)N_TX * HDIM * 2);
  __bf16* card1 = (__bf16*)carve((size_t)N_CARD * HDIM * 2);
  __bf16* mer1  = (__bf16*)carve((size_t)N_MER * HDIM * 2);
  __bf16* Wt_pl   = (__bf16*)carve(128 * 32 * 2);
  __bf16* Wt_rl   = (__bf16*)carve(128 * 32 * 2);
  __bf16* Wt_txr  = (__bf16*)carve(128 * 64 * 2);
  __bf16* Wt_rp_l = (__bf16*)carve(128 * 64 * 2);
  __bf16* Wt_rp_r = (__bf16*)carve(128 * 32 * 2);
  __bf16* Wt_rr_l = (__bf16*)carve(128 * 64 * 2);
  __bf16* Wt_rr_r = (__bf16*)carve(128 * 32 * 2);
  __bf16* Wt2c = (__bf16*)carve(128 * 128 * 2);
  __bf16* Wt2m = (__bf16*)carve(128 * 128 * 2);
  __bf16* Wt2t = (__bf16*)carve(128 * 128 * 2);
  float* b1tx = (float*)carve(128 * 4);
  float* b2c  = (float*)carve(128 * 4);

  hipMemsetAsync(d_ws, 0, zero_bytes, stream);

  prep_weights<<<128, 128, 0, stream>>>(
      l1_pays_wl, l1_pays_bl, l1_pays_wr,
      l1_recv_wl, l1_recv_bl, l1_recv_wr,
      l1_rpays_wl, l1_rpays_wr, l1_rrecv_wl, l1_rrecv_wr,
      w2l, b2, w2r,
      Wt_pl, Wt_rl, Wt_txr, Wt_rp_l, Wt_rp_r, Wt_rr_l, Wt_rr_r,
      Wt2c, Wt2m, Wt2t, b1tx, b2c);

  {
    long threads = (long)NEDGE * 16;
    int blocks = (int)((threads + 255) / 256);
    scatter_tx<<<blocks, 256, 0, stream>>>(x_tx, ps, pd, rs, rd,
                                           card_sum, cnt_card, mer_sum, cnt_mer);
  }

  {
    int ntiles = N_CARD / 16;  // 6250
    node_l1<<<(ntiles + 3) / 4, 128, 0, stream>>>(
        card_sum, cnt_card, x_card, Wt_rp_l, Wt_rp_r, l1_rpays_bl, card1, N_CARD);
  }
  {
    int ntiles = N_MER / 16;   // 625
    node_l1<<<(ntiles + 3) / 4, 128, 0, stream>>>(
        mer_sum, cnt_mer, x_mer, Wt_rr_l, Wt_rr_r, l1_rrecv_bl, mer1, N_MER);
  }
  {
    int ntiles = N_TX / 16;    // 62500
    tx_l1<<<ntiles / 4, 128, 0, stream>>>(
        x_card, x_mer, x_tx, ps, rs, Wt_pl, Wt_rl, Wt_txr, b1tx, tx1);
  }
  {
    int ntiles = N_TX / 16;    // 62500
    tx_l2_head<<<ntiles / 4, 128, 0, stream>>>(
        card1, mer1, tx1, ps, rs, Wt2c, Wt2m, Wt2t, b2c, wf, bf, (float*)d_out);
  }
}
